// NetLinkEvaluatePYG_86234353369872
// MI455X (gfx1250) — matrix operand order repaired and checked by reading, not compiled, not run on hardware
//
#include <hip/hip_runtime.h>
#include <hip/hip_bf16.h>

// ---------------------------------------------------------------------------
// MI455X (gfx1250) GCN link-predictor forward.
// Dominant cost: edge scatters (~2 GB traffic) -> coalesced gathers + f32
// atomics. GEMMs (~20 GFLOP) -> v_wmma_f32_16x16x32_bf16 (f32 accumulate),
// software-pipelined with LDS double buffering (1 barrier / k-step).
// ---------------------------------------------------------------------------

#define NFEAT 512
#define NHID  256

typedef __bf16 bf16_t;
typedef __attribute__((ext_vector_type(4)))  __bf16 v4bf;
typedef __attribute__((ext_vector_type(16))) __bf16 v16bf;
typedef __attribute__((ext_vector_type(8)))  float  v8f;

template <bool RELU>
static __device__ __forceinline__ v4bf cvt4_bf16(float4 v)
{
    if (RELU) {
        v.x = v.x > 0.f ? v.x : 0.f;
        v.y = v.y > 0.f ? v.y : 0.f;
        v.z = v.z > 0.f ? v.z : 0.f;
        v.w = v.w > 0.f ? v.w : 0.f;
    }
    v4bf r;
    r[0] = (bf16_t)v.x; r[1] = (bf16_t)v.y;
    r[2] = (bf16_t)v.z; r[3] = (bf16_t)v.w;
    return r;
}

// ---------------------------------------------------------------------------
// GEMM: C[M, NHID] = (RELU? relu(A) : A)[M, K] @ W[K, NHID]
// Block = 256 threads = 8 waves. Block tile = 128(M) x 64(N).
// Waves arranged 4(M) x 2(N); each wave owns a 32x32 tile = 4 WMMA tiles.
// Software pipeline: registers hold tile k+1 global loads while tile k runs
// through LDS fragments + WMMA. Double-buffered LDS, one barrier per k-step.
// ---------------------------------------------------------------------------
template <int K, bool RELU>
__global__ __launch_bounds__(256)
void gemm_bf16_wmma(const float* __restrict__ A, const float* __restrict__ W,
                    float* __restrict__ C, int M)
{
    // Row padded to 40 bf16 (80 B = 5*16 B) -> conflict-free aligned 16B reads.
    __shared__ __align__(16) bf16_t As[2][128][40];
    __shared__ __align__(16) bf16_t Bs[2][64][40];   // transposed: Bs[.][n][k]

    const int t  = threadIdx.x;
    const int w  = t >> 5;           // wave 0..7
    const int L  = t & 31;           // lane
    const int wm = w & 3;            // wave M index (0..3), 32 rows each
    const int wn = w >> 2;           // wave N index (0..1), 32 cols each
    const int r  = L & 15;
    const int hi = L >> 4;           // lane half

    const int m0 = blockIdx.x * 128;
    const int n0 = blockIdx.y * 64;

    // ---- Per-thread staging coordinates (invariant across k-steps). -----
    int aRow[4], aCol[4];
    const float* aPtr[4];
    #pragma unroll
    for (int i = 0; i < 4; ++i) {
        int f4  = t + 256 * i;               // float4 index, 8 per A row
        aRow[i] = f4 >> 3;
        aCol[i] = (f4 & 7) * 4;
        // Branch-free M guard: clamp row; bad rows only feed unwritten outputs.
        int gr = m0 + aRow[i]; gr = gr < (M - 1) ? gr : (M - 1);
        aPtr[i] = A + (size_t)gr * K + aCol[i];
    }
    int bKr[2], bNc[2];
    const float* bPtr[2];
    #pragma unroll
    for (int i = 0; i < 2; ++i) {
        int f4 = t + 256 * i;                // 16 float4 per W k-row of 64
        bKr[i] = f4 >> 4;
        bNc[i] = (f4 & 15) * 4;
        bPtr[i] = W + (size_t)bKr[i] * NHID + (n0 + bNc[i]);
    }

    v8f acc00 = {}, acc01 = {}, acc10 = {}, acc11 = {};

    // ---- Pipeline prologue: global loads for tile 0. --------------------
    float4 ra[4], rb[2];
    #pragma unroll
    for (int i = 0; i < 4; ++i) ra[i] = *(const float4*)aPtr[i];
    #pragma unroll
    for (int i = 0; i < 2; ++i) rb[i] = *(const float4*)bPtr[i];

    constexpr int NK = K / 32;
    for (int ki = 0; ki < NK; ++ki) {
        const int buf = ki & 1;

        // ---- Convert + store current tile into LDS buffer `buf`. --------
        #pragma unroll
        for (int i = 0; i < 4; ++i)
            *(v4bf*)&As[buf][aRow[i]][aCol[i]] = cvt4_bf16<RELU>(ra[i]);
        #pragma unroll
        for (int i = 0; i < 2; ++i) {
            v4bf b4 = cvt4_bf16<false>(rb[i]);
            Bs[buf][bNc[i] + 0][bKr[i]] = b4[0];
            Bs[buf][bNc[i] + 1][bKr[i]] = b4[1];
            Bs[buf][bNc[i] + 2][bKr[i]] = b4[2];
            Bs[buf][bNc[i] + 3][bKr[i]] = b4[3];
        }
        __syncthreads();   // stores of `buf` visible; prev frag reads done

        // ---- Issue next tile's global loads (hidden behind WMMA). -------
        if (ki + 1 < NK) {
            const int kt = (ki + 1) * 32;
            #pragma unroll
            for (int i = 0; i < 4; ++i)
                ra[i] = *(const float4*)(aPtr[i] + kt);
            #pragma unroll
            for (int i = 0; i < 2; ++i)
                rb[i] = *(const float4*)(bPtr[i] + (size_t)kt * NHID);
        }

        // ---- Fragments --------------------------------------------------
        // A (16x32): lane holds row r; elem e -> K = (e<8 ? e : e+8) + 8*hi
        // B (32x16): lane holds col r; elem e -> K = (e<8 ? e : e+8) + 8*hi
        v16bf a0, a1, b0, b1;
        const bf16_t* rowA0 = &As[buf][wm * 32 + r][0];
        const bf16_t* rowA1 = &As[buf][wm * 32 + 16 + r][0];
        const bf16_t* rowB0 = &Bs[buf][wn * 32 + r][0];
        const bf16_t* rowB1 = &Bs[buf][wn * 32 + 16 + r][0];
        #pragma unroll
        for (int e = 0; e < 8; ++e) {
            a0[e]     = rowA0[hi * 8 + e];
            a0[8 + e] = rowA0[16 + hi * 8 + e];
            a1[e]     = rowA1[hi * 8 + e];
            a1[8 + e] = rowA1[16 + hi * 8 + e];
        }
        #pragma unroll
        for (int e = 0; e < 8; ++e) {
            b0[e]     = rowB0[hi * 8 + e];
            b0[8 + e] = rowB0[16 + hi * 8 + e];
            b1[e]     = rowB1[hi * 8 + e];
            b1[8 + e] = rowB1[16 + hi * 8 + e];
        }

        acc00 = __builtin_amdgcn_wmma_f32_16x16x32_bf16(false, a0, false, b0,
                                                        (short)0, acc00, false, false);
        acc01 = __builtin_amdgcn_wmma_f32_16x16x32_bf16(false, a0, false, b1,
                                                        (short)0, acc01, false, false);
        acc10 = __builtin_amdgcn_wmma_f32_16x16x32_bf16(false, a1, false, b0,
                                                        (short)0, acc10, false, false);
        acc11 = __builtin_amdgcn_wmma_f32_16x16x32_bf16(false, a1, false, b1,
                                                        (short)0, acc11, false, false);
        // No trailing barrier: next iteration writes the *other* LDS buffer,
        // and the next top-of-loop barrier orders it against these reads.
    }

    // ---- Epilogue: D layout VGPR v -> M = v + 8*hi, N = lane&15. --------
    const int cb = n0 + wn * 32 + r;
    #pragma unroll
    for (int v = 0; v < 8; ++v) {
        int row0 = m0 + wm * 32 + v + 8 * hi;        // mi = 0
        int row1 = row0 + 16;                        // mi = 1
        if (row0 < M) {
            C[(size_t)row0 * NHID + cb]      = acc00[v];
            C[(size_t)row0 * NHID + cb + 16] = acc01[v];
        }
        if (row1 < M) {
            C[(size_t)row1 * NHID + cb]      = acc10[v];
            C[(size_t)row1 * NHID + cb + 16] = acc11[v];
        }
    }
}

// ---------------------------------------------------------------------------
__global__ __launch_bounds__(256)
void zero_kernel(float4* __restrict__ p, int n4)
{
    int i = blockIdx.x * blockDim.x + threadIdx.x;
    if (i < n4) p[i] = make_float4(0.f, 0.f, 0.f, 0.f);
}

// One block per edge; lane f handles feature f. Coalesced gather of H[src],
// fire-and-forget global_atomic_add_f32 into Z[dst].
__global__ __launch_bounds__(256)
void scatter_add(const float* __restrict__ H, const int* __restrict__ src,
                 const int* __restrict__ dst, const float* __restrict__ ew,
                 float* __restrict__ Z, int E)
{
    int e = blockIdx.x;
    if (e >= E) return;
    int   s = src[e];
    int   d = dst[e];
    float w = ew[e];
    int   f = threadIdx.x;
    (void)atomicAdd(&Z[(size_t)d * NHID + f], w * H[(size_t)s * NHID + f]);
}

// ---------------------------------------------------------------------------
// Decode: out[p, j] = sum_k Z[ps][k]*Wlin[k][j] + Z[pd][k]*Wlin[256+k][j]
// One wave32 per positive edge; lane-strided accumulation + shuffle reduce.
// ---------------------------------------------------------------------------
__global__ __launch_bounds__(256)
void decode_kernel(const float* __restrict__ Z, const int* __restrict__ ps,
                   const int* __restrict__ pd, const float* __restrict__ Wlin,
                   float* __restrict__ out, int P)
{
    int wave = (int)((blockIdx.x * (size_t)blockDim.x + threadIdx.x) >> 5);
    int lane = threadIdx.x & 31;
    if (wave >= P) return;

    const float* zs = Z + (size_t)ps[wave] * NHID;
    const float* zd = Z + (size_t)pd[wave] * NHID;

    float a0 = 0.f, a1 = 0.f;
    #pragma unroll 4
    for (int k = lane; k < NHID; k += 32) {
        float vs = zs[k], vd = zd[k];
        a0 += vs * Wlin[2 * k]     + vd * Wlin[2 * (NHID + k)];
        a1 += vs * Wlin[2 * k + 1] + vd * Wlin[2 * (NHID + k) + 1];
    }
    #pragma unroll
    for (int off = 16; off > 0; off >>= 1) {
        a0 += __shfl_down(a0, off, 32);
        a1 += __shfl_down(a1, off, 32);
    }
    if (lane == 0) {
        out[2 * (size_t)wave]     = a0;
        out[2 * (size_t)wave + 1] = a1;
    }
}

// ---------------------------------------------------------------------------
extern "C" void kernel_launch(void* const* d_in, const int* in_sizes, int n_in,
                              void* d_out, int out_size, void* d_ws, size_t ws_size,
                              hipStream_t stream)
{
    const float* x   = (const float*)d_in[0];   // [M, 512]
    const int*   ei  = (const int*)  d_in[1];   // [2, E]
    const float* ew  = (const float*)d_in[2];   // [E]
    const int*   pei = (const int*)  d_in[3];   // [2, P]
    const float* W1  = (const float*)d_in[4];   // [512, 256]
    const float* W2  = (const float*)d_in[5];   // [256, 256]
    const float* Wl  = (const float*)d_in[6];   // [512, 2]
    float*       out = (float*)d_out;           // [P, 2]

    const int M = in_sizes[0] / NFEAT;          // 50000
    const int E = in_sizes[2];                  // 800000
    const int P = in_sizes[3] / 2;              // 200000

    // Workspace: H (M x 256, reused for H1 then H2), Z (M x 256, Z1 then Z2).
    float* H = (float*)d_ws;
    float* Z = H + (size_t)M * NHID;

    const dim3 gemm_grid((M + 127) / 128, NHID / 64);
    const int  n4 = (M * NHID) / 4;
    const int  zero_blocks = (n4 + 255) / 256;
    const int  dec_blocks  = (P + 7) / 8;       // 8 waves/block, 1 wave/edge

    // Layer 1: H1 = X @ W1
    gemm_bf16_wmma<NFEAT, false><<<gemm_grid, 256, 0, stream>>>(x, W1, H, M);

    // Z1 = segment_sum(ew * H1[src] -> dst)
    zero_kernel<<<zero_blocks, 256, 0, stream>>>((float4*)Z, n4);
    scatter_add<<<E, 256, 0, stream>>>(H, ei, ei + E, ew, Z, E);

    // Layer 2: H2 = relu(Z1) @ W2   (ReLU fused into A staging)
    gemm_bf16_wmma<NHID, true><<<gemm_grid, 256, 0, stream>>>(Z, W2, H, M);

    // Z2 = segment_sum(ew * H2[src] -> dst)
    zero_kernel<<<zero_blocks, 256, 0, stream>>>((float4*)Z, n4);
    scatter_add<<<E, 256, 0, stream>>>(H, ei, ei + E, ew, Z, E);

    // Decode positive edges.
    decode_kernel<<<dec_blocks, 256, 0, stream>>>(Z, pei, pei + P, Wl, out, P);
}
